// EncoderLayers_44074954391812
// MI455X (gfx1250) — matrix operand order repaired and checked by reading, not compiled, not run on hardware
//
#include <hip/hip_runtime.h>
#include <cstdint>
#include <cstddef>

// ---------------------------------------------------------------------------
// MI455X (gfx1250) encoder layer. Wave32, WMMA bf16 16x16x32, f32 accum.
// Main GEMMs: double-buffered LDS fed by global_load_async_to_lds_b128
// (ASYNCcnt-tracked), 8 WMMA per wave per 32-K step. Attention scores kept
// in f16 to halve the dominant HBM round-trip; attention GEMMs batched over
// the 4-sample batch via gridDim.z.
// ---------------------------------------------------------------------------

typedef __attribute__((ext_vector_type(16))) __bf16 v16bf;
typedef __attribute__((ext_vector_type(8)))  __bf16 v8bf;
typedef __attribute__((ext_vector_type(8)))  float  v8f;

#define BATCH 4
#define SEQ   2048
#define DMODEL 1024
#define NHEAD 16
#define HDIM  64
#define FFN1  4096
#define FFN2  2048
#define FFN3  1024

#define BM 128
#define BK 32
#define LDS_STRIDE 40   // padded row stride in bf16 elems (80B, 16B multiple)

// Output modes for the GEMM epilogue
#define OUT_F32  0
#define OUT_BF16 1
#define OUT_F16  2

// 16-byte global -> LDS async copy (per lane). VDST = LDS byte offset
// (generic LDS address truncated to 32 bits per aperture encoding).
__device__ __forceinline__ void async_cp16(const void* lds_dst,
                                           const void* g_src)
{
    uint32_t l = (uint32_t)(uintptr_t)lds_dst;
    unsigned long long g = (unsigned long long)(uintptr_t)g_src;
    asm volatile("global_load_async_to_lds_b128 %0, %1, off"
                 :: "v"(l), "v"(g) : "memory");
}
__device__ __forceinline__ void wait_async0()
{
    asm volatile("s_wait_asynccnt 0x0" ::: "memory");
}

// ---------------------------------------------------------------------------
// C[M,N] = alpha * A[M,K] @ B[N,K]^T + bias, optional ReLU, batched over
// gridDim.z with element strides strA/strB/strC.
// Block tile BM=128 x BNT, 256 threads (8 waves).
//   BNT=128: waves 2(M) x 4(N), wave tile 64x32 -> 4x2 WMMA.
// Double-buffered LDS, async global->LDS copies overlapped with WMMA.
// M % BM == 0, N % BNT == 0, K % BK == 0 (guaranteed by the call sites).
// ---------------------------------------------------------------------------
template <int BNT, int OUTM, bool RELU>
__global__ __launch_bounds__(256)
void gemm_nk_async(const __bf16* __restrict__ A, int lda, size_t strA,
                   const __bf16* __restrict__ B, int ldb, size_t strB,
                   void* __restrict__ C, int ldc, size_t strC,
                   const float* __restrict__ bias, float alpha, int K)
{
    constexpr int WAVES_M = (BNT == 128) ? 2 : 4;
    constexpr int WAVES_N = 8 / WAVES_M;
    constexpr int MI  = BM  / (16 * WAVES_M);   // A frags per wave
    constexpr int NJ  = BNT / (16 * WAVES_N);   // B frags per wave
    constexpr int ACH = (BM  * BK) / (8 * 256); // 16B chunks/thread for A
    constexpr int BCH = (BNT * BK) / (8 * 256); // 16B chunks/thread for B

    __shared__ __align__(16) __bf16 As[2][BM  * LDS_STRIDE];
    __shared__ __align__(16) __bf16 Bs[2][BNT * LDS_STRIDE];

    const int tid  = threadIdx.x;
    const int lane = tid & 31;
    const int wave = tid >> 5;
    const int wm   = wave % WAVES_M;
    const int wn   = wave / WAVES_M;
    const int m0   = blockIdx.y * BM;
    const int n0   = blockIdx.x * BNT;
    const int hl   = lane >> 4;
    const int l15  = lane & 15;

    A += (size_t)blockIdx.z * strA;
    B += (size_t)blockIdx.z * strB;
    const size_t zc = (size_t)blockIdx.z * strC;

    auto stage = [&](int k0, int p) {
#pragma unroll
        for (int it = 0; it < ACH; ++it) {
            int idx = tid + it * 256;
            int row = idx >> 2;
            int ck  = (idx & 3) * 8;
            async_cp16(&As[p][row * LDS_STRIDE + ck],
                       A + (size_t)(m0 + row) * lda + k0 + ck);
        }
#pragma unroll
        for (int it = 0; it < BCH; ++it) {
            int idx = tid + it * 256;
            int row = idx >> 2;
            int ck  = (idx & 3) * 8;
            async_cp16(&Bs[p][row * LDS_STRIDE + ck],
                       B + (size_t)(n0 + row) * ldb + k0 + ck);
        }
    };

    v8f acc[MI][NJ] = {};

    stage(0, 0);
    wait_async0();
    __syncthreads();

    int p = 0;
    for (int k0 = 0; k0 < K; k0 += BK) {
        if (k0 + BK < K) stage(k0 + BK, p ^ 1);   // prefetch next tile

        // A fragment (16x32 bf16): half 0 -> K {0..7,16..23}, half 1 -> +8
        v16bf afr[MI];
#pragma unroll
        for (int i = 0; i < MI; ++i) {
            int r  = (wm * MI + i) * 16 + l15;
            int kb = hl * 8;
            v8bf lo = *(const v8bf*)&As[p][r * LDS_STRIDE + kb];
            v8bf hi = *(const v8bf*)&As[p][r * LDS_STRIDE + kb + 16];
            afr[i] = __builtin_shufflevector(lo, hi,
                     0,1,2,3,4,5,6,7,8,9,10,11,12,13,14,15);
        }
        // B fragment (32x16 bf16): half 0 -> K {0..7,16..23}, half 1 -> +8
        v16bf bfr[NJ];
#pragma unroll
        for (int j = 0; j < NJ; ++j) {
            int c  = (wn * NJ + j) * 16 + l15;
            int kb = hl * 8;
            v8bf lo = *(const v8bf*)&Bs[p][c * LDS_STRIDE + kb];
            v8bf hi = *(const v8bf*)&Bs[p][c * LDS_STRIDE + kb + 16];
            bfr[j] = __builtin_shufflevector(lo, hi,
                     0,1,2,3,4,5,6,7,8,9,10,11,12,13,14,15);
        }
#pragma unroll
        for (int i = 0; i < MI; ++i)
#pragma unroll
            for (int j = 0; j < NJ; ++j)
                acc[i][j] = __builtin_amdgcn_wmma_f32_16x16x32_bf16(
                    false, afr[i], false, bfr[j], (short)0, acc[i][j],
                    false, false);

        wait_async0();      // next tile landed in LDS
        __syncthreads();    // ... and visible to all waves
        p ^= 1;
    }

    // epilogue: D layout = 8 VGPRs, row = r + 8*hl, col = l15
#pragma unroll
    for (int i = 0; i < MI; ++i) {
#pragma unroll
        for (int j = 0; j < NJ; ++j) {
            int n = n0 + (wn * NJ + j) * 16 + l15;
            float bv = bias ? bias[n] : 0.0f;
#pragma unroll
            for (int r = 0; r < 8; ++r) {
                int m = m0 + (wm * MI + i) * 16 + hl * 8 + r;
                float v = acc[i][j][r] * alpha + bv;
                if (RELU) v = fmaxf(v, 0.0f);
                if (OUTM == OUT_BF16)
                    ((__bf16*)C)[zc + (size_t)m * ldc + n] = (__bf16)v;
                else if (OUTM == OUT_F16)
                    ((_Float16*)C)[zc + (size_t)m * ldc + n] = (_Float16)v;
                else
                    ((float*)C)[zc + (size_t)m * ldc + n] = v;
            }
        }
    }
}

// ---------------------------------------------------------------------------
// A.V GEMM: C[M,64] = A[M,K] @ B[K,64]  (B row-major [K,N], transposed while
// staging through LDS). bf16 out. Software-pipelined register staging,
// batched over gridDim.z. M%128==0, K%32==0, N==64.
// ---------------------------------------------------------------------------
__global__ __launch_bounds__(256)
void gemm_kn_av(const __bf16* __restrict__ A, int lda, size_t strA,
                const __bf16* __restrict__ B, int ldb, size_t strB,
                __bf16* __restrict__ C, int ldc, size_t strC, int K)
{
    __shared__ __align__(16) __bf16 As[BM * LDS_STRIDE];
    __shared__ __align__(16) __bf16 Bs[64 * LDS_STRIDE];

    const int tid  = threadIdx.x;
    const int lane = tid & 31;
    const int wave = tid >> 5;
    const int wm   = wave & 3;
    const int wn   = wave >> 2;
    const int m0   = blockIdx.y * BM;
    const int hl   = lane >> 4;
    const int l15  = lane & 15;

    A += (size_t)blockIdx.z * strA;
    B += (size_t)blockIdx.z * strB;
    C += (size_t)blockIdx.z * strC;

    const int arow0 = tid >> 2;            // A staging coords
    const int ack   = (tid & 3) * 8;
    const int bkk   = tid >> 3;            // B staging: k row 0..31
    const int bnb   = (tid & 7) * 8;       // n chunk base

    v8f acc[2][2] = {};
    v8bf ra[2], rb;

#pragma unroll
    for (int it = 0; it < 2; ++it)
        ra[it] = *(const v8bf*)(A + (size_t)(m0 + arow0 + it * 64) * lda + ack);
    rb = *(const v8bf*)(B + (size_t)bkk * ldb + bnb);

    for (int k0 = 0; k0 < K; k0 += BK) {
        __syncthreads();   // everyone done reading previous tile
#pragma unroll
        for (int it = 0; it < 2; ++it)
            *(v8bf*)&As[(arow0 + it * 64) * LDS_STRIDE + ack] = ra[it];
#pragma unroll
        for (int j = 0; j < 8; ++j)        // transpose into [N][K]
            Bs[(bnb + j) * LDS_STRIDE + bkk] = rb[j];
        __syncthreads();

        if (k0 + BK < K) {                 // issue next tile's loads now;
#pragma unroll                             // waits land after the WMMAs
            for (int it = 0; it < 2; ++it)
                ra[it] = *(const v8bf*)(A +
                    (size_t)(m0 + arow0 + it * 64) * lda + (k0 + BK) + ack);
            rb = *(const v8bf*)(B + (size_t)(k0 + BK + bkk) * ldb + bnb);
            __builtin_prefetch(A + (size_t)(m0 + arow0) * lda + k0 + 2 * BK,
                               0, 1);
        }

        v16bf afr[2], bfr[2];
#pragma unroll
        for (int i = 0; i < 2; ++i) {
            int r  = wm * 32 + i * 16 + l15;
            int kb = hl * 8;
            v8bf lo = *(const v8bf*)&As[r * LDS_STRIDE + kb];
            v8bf hi = *(const v8bf*)&As[r * LDS_STRIDE + kb + 16];
            afr[i] = __builtin_shufflevector(lo, hi,
                     0,1,2,3,4,5,6,7,8,9,10,11,12,13,14,15);
        }
#pragma unroll
        for (int j = 0; j < 2; ++j) {
            int c  = wn * 32 + j * 16 + l15;
            int kb = hl * 8;
            v8bf lo = *(const v8bf*)&Bs[c * LDS_STRIDE + kb];
            v8bf hi = *(const v8bf*)&Bs[c * LDS_STRIDE + kb + 16];
            bfr[j] = __builtin_shufflevector(lo, hi,
                     0,1,2,3,4,5,6,7,8,9,10,11,12,13,14,15);
        }
#pragma unroll
        for (int i = 0; i < 2; ++i)
#pragma unroll
            for (int j = 0; j < 2; ++j)
                acc[i][j] = __builtin_amdgcn_wmma_f32_16x16x32_bf16(
                    false, afr[i], false, bfr[j], (short)0, acc[i][j],
                    false, false);
    }

#pragma unroll
    for (int i = 0; i < 2; ++i)
#pragma unroll
        for (int j = 0; j < 2; ++j) {
            int n = wn * 32 + j * 16 + l15;
#pragma unroll
            for (int r = 0; r < 8; ++r) {
                int m = m0 + wm * 32 + i * 16 + hl * 8 + r;
                C[(size_t)m * ldc + n] = (__bf16)acc[i][j][r];
            }
        }
}

// ---------------------------------------------------------------------------
// Softmax over last dim with additive mask, NaN->0, summed over batch.
// Scores arrive as f16 (halves HBM traffic of the dominant buffer).
// ---------------------------------------------------------------------------
__global__ __launch_bounds__(256)
void softmax_batchsum(const _Float16* __restrict__ scores, // [B][S][S]
                      const float* __restrict__ mask,      // [S][S]
                      __bf16* __restrict__ Ah)             // [S][S]
{
    __shared__ float red[256];
    const int s   = blockIdx.x;
    const int tid = threadIdx.x;
    float accv[8] = {0, 0, 0, 0, 0, 0, 0, 0};
    const float* mrow = mask + (size_t)s * SEQ;

    for (int b = 0; b < BATCH; ++b) {
        const _Float16* row = scores + ((size_t)b * SEQ + s) * SEQ;
        float x[8];
        float mx = -3.402823466e38f;
#pragma unroll
        for (int j = 0; j < 8; ++j) {
            int c = tid + j * 256;
            x[j]  = (float)row[c] + mrow[c];
            mx    = fmaxf(mx, x[j]);
        }
        red[tid] = mx; __syncthreads();
        for (int st = 128; st > 0; st >>= 1) {
            if (tid < st) red[tid] = fmaxf(red[tid], red[tid + st]);
            __syncthreads();
        }
        mx = red[0]; __syncthreads();

        float sum = 0.0f;
#pragma unroll
        for (int j = 0; j < 8; ++j) { x[j] = __expf(x[j] - mx); sum += x[j]; }
        red[tid] = sum; __syncthreads();
        for (int st = 128; st > 0; st >>= 1) {
            if (tid < st) red[tid] += red[tid + st];
            __syncthreads();
        }
        sum = red[0]; __syncthreads();

        float inv = (sum > 0.0f) ? 1.0f / sum : 0.0f;   // NaN -> 0
#pragma unroll
        for (int j = 0; j < 8; ++j) accv[j] += x[j] * inv;
    }
#pragma unroll
    for (int j = 0; j < 8; ++j)
        Ah[(size_t)s * SEQ + tid + j * 256] = (__bf16)accv[j];
}

// ---------------------------------------------------------------------------
// Fused residual add + LayerNorm over D=1024; one workgroup per row.
// ---------------------------------------------------------------------------
__global__ __launch_bounds__(256)
void add_layernorm(const float* __restrict__ a, const float* __restrict__ b,
                   const float* __restrict__ gamma,
                   const float* __restrict__ beta,
                   float* __restrict__ out_f32,       // may be nullptr
                   __bf16* __restrict__ out_bf16)
{
    __shared__ float r1[256];
    __shared__ float r2[256];
    const int row = blockIdx.x;
    const int tid = threadIdx.x;
    const size_t base = (size_t)row * DMODEL;

    float v[4];
    float s = 0.0f, ss = 0.0f;
#pragma unroll
    for (int j = 0; j < 4; ++j) {
        int c = tid + j * 256;
        v[j]  = a[base + c] + b[base + c];
        s    += v[j];
        ss   += v[j] * v[j];
    }
    r1[tid] = s; r2[tid] = ss; __syncthreads();
    for (int st = 128; st > 0; st >>= 1) {
        if (tid < st) { r1[tid] += r1[tid + st]; r2[tid] += r2[tid + st]; }
        __syncthreads();
    }
    float mean = r1[0] * (1.0f / DMODEL);
    float var  = r2[0] * (1.0f / DMODEL) - mean * mean;
    float rstd = rsqrtf(var + 1e-5f);

#pragma unroll
    for (int j = 0; j < 4; ++j) {
        int c   = tid + j * 256;
        float y = gamma[c] * ((v[j] - mean) * rstd) + beta[c];
        if (out_f32) out_f32[base + c] = y;
        out_bf16[base + c] = (__bf16)y;
    }
}

__global__ __launch_bounds__(256)
void cvt_f32_bf16(const float* __restrict__ in, __bf16* __restrict__ out,
                  size_t n)
{
    size_t i      = (size_t)blockIdx.x * blockDim.x + threadIdx.x;
    size_t stride = (size_t)gridDim.x * blockDim.x;
    for (; i < n; i += stride) out[i] = (__bf16)in[i];
}

// ---------------------------------------------------------------------------
// Host orchestration
// ---------------------------------------------------------------------------
extern "C" void kernel_launch(void* const* d_in, const int* in_sizes, int n_in,
                              void* d_out, int out_size, void* d_ws,
                              size_t ws_size, hipStream_t stream)
{
    (void)in_sizes; (void)n_in; (void)out_size; (void)ws_size;

    const float* x     = (const float*)d_in[0];
    const float* mask  = (const float*)d_in[1];
    const float* Wq    = (const float*)d_in[2];
    const float* bq    = (const float*)d_in[3];
    const float* Wk    = (const float*)d_in[4];
    const float* bk    = (const float*)d_in[5];
    const float* Wv    = (const float*)d_in[6];
    const float* bv    = (const float*)d_in[7];
    const float* Wo    = (const float*)d_in[8];
    const float* bo    = (const float*)d_in[9];
    const float* gamma = (const float*)d_in[10];
    const float* beta  = (const float*)d_in[11];
    const float* W1    = (const float*)d_in[12];
    const float* b1    = (const float*)d_in[13];
    const float* W2    = (const float*)d_in[14];
    const float* b2    = (const float*)d_in[15];
    const float* W3    = (const float*)d_in[16];
    const float* b3    = (const float*)d_in[17];
    const float* W4    = (const float*)d_in[18];
    const float* b4    = (const float*)d_in[19];
    const float* Wl    = (const float*)d_in[20];
    const float* bl    = (const float*)d_in[21];

    const size_t NTOK = (size_t)BATCH * SEQ;
    const size_t NX   = NTOK * DMODEL;
    const size_t SBD  = (size_t)SEQ * DMODEL;   // per-batch activation stride
    const size_t SS   = (size_t)SEQ * SEQ;

    char*  ws  = (char*)d_ws;
    size_t off = 0;
    auto take = [&](size_t bytes) -> char* {
        char* p = ws + off;
        off += (bytes + 255) & ~(size_t)255;
        return p;
    };
    __bf16*   xb   = (__bf16*)  take(NX * 2);
    __bf16*   Wqb  = (__bf16*)  take((size_t)DMODEL * DMODEL * 2);
    __bf16*   Wkb  = (__bf16*)  take((size_t)DMODEL * DMODEL * 2);
    __bf16*   Wvb  = (__bf16*)  take((size_t)DMODEL * DMODEL * 2);
    __bf16*   Wob  = (__bf16*)  take((size_t)DMODEL * DMODEL * 2);
    __bf16*   W1b  = (__bf16*)  take((size_t)FFN1 * DMODEL * 2);
    __bf16*   W2b  = (__bf16*)  take((size_t)FFN2 * FFN1 * 2);
    __bf16*   W3b  = (__bf16*)  take((size_t)FFN3 * FFN2 * 2);
    __bf16*   W4b  = (__bf16*)  take((size_t)DMODEL * FFN3 * 2);
    __bf16*   Wlb  = (__bf16*)  take((size_t)DMODEL * DMODEL * 2);
    __bf16*   Qb   = (__bf16*)  take(NX * 2);
    __bf16*   Kb   = (__bf16*)  take(NX * 2);
    __bf16*   Vb   = (__bf16*)  take(NX * 2);
    _Float16* scr  = (_Float16*)take((size_t)BATCH * SS * 2);  // f16 scores
    __bf16*   Ah   = (__bf16*)  take(SS * 2);
    __bf16*   Vcat = (__bf16*)  take(NX * 2);
    float*    attn = (float*)   take(NX * 4);
    float*    ln1f = (float*)   take(NX * 4);
    __bf16*   ln1b = (__bf16*)  take(NX * 2);
    __bf16*   f1b  = (__bf16*)  take(NTOK * FFN1 * 2);
    __bf16*   f2b  = (__bf16*)  take(NTOK * FFN2 * 2);
    __bf16*   f3b  = (__bf16*)  take(NTOK * FFN3 * 2);
    float*    f4f  = (float*)   take(NX * 4);
    __bf16*   ln2b = (__bf16*)  take(NX * 2);

    auto cvt = [&](const float* s, __bf16* d, size_t n) {
        cvt_f32_bf16<<<1024, 256, 0, stream>>>(s, d, n);
    };
    cvt(x,  xb,  NX);
    cvt(Wq, Wqb, (size_t)DMODEL * DMODEL);
    cvt(Wk, Wkb, (size_t)DMODEL * DMODEL);
    cvt(Wv, Wvb, (size_t)DMODEL * DMODEL);
    cvt(Wo, Wob, (size_t)DMODEL * DMODEL);
    cvt(W1, W1b, (size_t)FFN1 * DMODEL);
    cvt(W2, W2b, (size_t)FFN2 * FFN1);
    cvt(W3, W3b, (size_t)FFN3 * FFN2);
    cvt(W4, W4b, (size_t)DMODEL * FFN3);
    cvt(Wl, Wlb, (size_t)DMODEL * DMODEL);

    const dim3 blk(256);
    auto grid = [](size_t M, size_t N, unsigned Z = 1) {
        return dim3((unsigned)(N / 128), (unsigned)(M / BM), Z);
    };

    // Q, K, V projections -> bf16
    gemm_nk_async<128, OUT_BF16, false><<<grid(NTOK, DMODEL), blk, 0, stream>>>(
        xb, DMODEL, 0, Wqb, DMODEL, 0, Qb, DMODEL, 0, bq, 1.0f, DMODEL);
    gemm_nk_async<128, OUT_BF16, false><<<grid(NTOK, DMODEL), blk, 0, stream>>>(
        xb, DMODEL, 0, Wkb, DMODEL, 0, Kb, DMODEL, 0, bk, 1.0f, DMODEL);
    gemm_nk_async<128, OUT_BF16, false><<<grid(NTOK, DMODEL), blk, 0, stream>>>(
        xb, DMODEL, 0, Wvb, DMODEL, 0, Vb, DMODEL, 0, bv, 1.0f, DMODEL);

    // attention per head, batch handled by gridDim.z
    const float inv_sqrt_d = 0.125f;
    for (int h = 0; h < NHEAD; ++h) {
        const size_t ho = (size_t)h * HDIM;
        // scores[b] = (Q_h K_h^T) / 8  -> f16
        gemm_nk_async<128, OUT_F16, false>
            <<<grid(SEQ, SEQ, BATCH), blk, 0, stream>>>(
                Qb + ho, DMODEL, SBD, Kb + ho, DMODEL, SBD,
                scr, SEQ, SS, nullptr, inv_sqrt_d, HDIM);
        // A_h = sum_b softmax(scores[b] + mask) -> bf16
        softmax_batchsum<<<SEQ, blk, 0, stream>>>(scr, mask, Ah);
        // V_out[b,:,h*64:(h+1)*64] = A_h @ V_h[b]
        gemm_kn_av<<<dim3(1, SEQ / BM, BATCH), blk, 0, stream>>>(
            Ah, SEQ, 0, Vb + ho, DMODEL, SBD, Vcat + ho, DMODEL, SBD, SEQ);
    }

    // output projection -> f32
    gemm_nk_async<128, OUT_F32, false><<<grid(NTOK, DMODEL), blk, 0, stream>>>(
        Vcat, DMODEL, 0, Wob, DMODEL, 0, attn, DMODEL, 0, bo, 1.0f, DMODEL);

    // LN1
    add_layernorm<<<NTOK, blk, 0, stream>>>(attn, x, gamma, beta, ln1f, ln1b);

    // FFN
    gemm_nk_async<128, OUT_BF16, true><<<grid(NTOK, FFN1), blk, 0, stream>>>(
        ln1b, DMODEL, 0, W1b, DMODEL, 0, f1b, FFN1, 0, b1, 1.0f, DMODEL);
    gemm_nk_async<128, OUT_BF16, true><<<grid(NTOK, FFN2), blk, 0, stream>>>(
        f1b, FFN1, 0, W2b, FFN1, 0, f2b, FFN2, 0, b2, 1.0f, FFN1);
    gemm_nk_async<128, OUT_BF16, true><<<grid(NTOK, FFN3), blk, 0, stream>>>(
        f2b, FFN2, 0, W3b, FFN2, 0, f3b, FFN3, 0, b3, 1.0f, FFN2);
    gemm_nk_async<128, OUT_F32, false><<<grid(NTOK, DMODEL), blk, 0, stream>>>(
        f3b, FFN3, 0, W4b, FFN3, 0, f4f, DMODEL, 0, b4, 1.0f, FFN3);

    // LN2
    add_layernorm<<<NTOK, blk, 0, stream>>>(ln1f, f4f, gamma, beta,
                                            nullptr, ln2b);

    // final linear -> d_out
    gemm_nk_async<128, OUT_F32, false><<<grid(NTOK, DMODEL), blk, 0, stream>>>(
        ln2b, DMODEL, 0, Wlb, DMODEL, 0, (float*)d_out, DMODEL, 0,
        bl, 1.0f, DMODEL);
}
